// Decoder_45165876085308
// MI455X (gfx1250) — matrix operand order repaired and checked by reading, not compiled, not run on hardware
//
#include <hip/hip_runtime.h>
#include <hip/hip_bf16.h>
#include <math.h>

// ---------------------------------------------------------------------------
// Shapes (fixed by the reference)
// ---------------------------------------------------------------------------
#define VSZ 32000
#define ESZ 512
#define HSZ 512
#define ENC 1024
#define SSZ 128
#define BSZ 64
#define KOUT 2048            // H + ENC + E

typedef __attribute__((ext_vector_type(16))) __bf16          v16bf;
typedef __attribute__((ext_vector_type(16))) unsigned short  v16us;
typedef __attribute__((ext_vector_type(8)))  float           v8f;

#if __has_builtin(__builtin_amdgcn_global_load_async_to_lds_b128) && \
    __has_builtin(__builtin_amdgcn_s_wait_asynccnt)
#define HAVE_ASYNC_LDS 1
// exact parameter types from the round-2 diagnostic: <4 x i32> in AS1 / AS3
typedef int v4i __attribute__((vector_size(16)));
typedef __attribute__((address_space(1))) v4i* gv4i_p;
typedef __attribute__((address_space(3))) v4i* lv4i_p;
#define ASYNC_COPY16(g, l)                                                    \
  __builtin_amdgcn_global_load_async_to_lds_b128((gv4i_p)(g), (lv4i_p)(l), 0, 0)
#define WAIT_ASYNC_N(imm) __builtin_amdgcn_s_wait_asynccnt(imm)
#else
#define HAVE_ASYNC_LDS 0
#define ASYNC_COPY16(g, l) (void)0
#define WAIT_ASYNC_N(imm) (void)0
#endif

// ---------------------------------------------------------------------------
// Fragment loaders (contiguous k = 16h + e per lane half, wave32)
// ---------------------------------------------------------------------------

// A-tile 16x32 (MxK) from row-major f32, converted with native bf16 cvt.
// lanes 0-15: M=lane, K=kb..kb+15 with kb=0; lanes 16-31: kb=16.
__device__ __forceinline__ v16bf load_a_f32(const float* A, int ld, int k0) {
  int lane = threadIdx.x & 31;
  int m    = lane & 15;
  int kb   = (lane < 16) ? 0 : 16;
  const float* p = A + (size_t)m * ld + k0 + kb;
  v16bf u;
#pragma unroll
  for (int e = 0; e < 8; ++e) u[e]     = (__bf16)p[e];
#pragma unroll
  for (int e = 0; e < 8; ++e) u[8 + e] = (__bf16)p[8 + e];
  return u;
}

// A-tile 16x32 (MxK) from row-major bf16 (pre-converted): pure vector loads.
__device__ __forceinline__ v16bf load_a_bf16(const unsigned short* A, int ld,
                                             int k0) {
  int lane = threadIdx.x & 31;
  int m    = lane & 15;
  int kb   = (lane < 16) ? 0 : 16;
  const unsigned short* p = A + (size_t)m * ld + k0 + kb;
  v16us u;
#pragma unroll
  for (int e = 0; e < 8; ++e) u[e]     = p[e];
#pragma unroll
  for (int e = 0; e < 8; ++e) u[8 + e] = p[8 + e];
  return __builtin_bit_cast(v16bf, u);
}

// B-tile 32x16 (KxN) where the source is stored N-major as rows of K
// (i.e. Wrows[n][k]); lane = N, elements e -> K = kb+e, kb = 0/16 per half.
// Each lane reads 16 *contiguous* bf16 (32B).
__device__ __forceinline__ v16bf load_b_bf16rows(const unsigned short* W,
                                                 int ld, int k0, int n0) {
  int lane = threadIdx.x & 31;
  int n    = n0 + (lane & 15);
  int kb   = k0 + ((lane < 16) ? 0 : 16);
  const unsigned short* p = W + (size_t)n * ld + kb;
  v16us u;
#pragma unroll
  for (int e = 0; e < 16; ++e) u[e] = p[e];
  return __builtin_bit_cast(v16bf, u);
}

// ---------------------------------------------------------------------------
// K-1: transpose+convert att_We [ENC x H] f32 -> Wt [H x ENC] bf16 (2MB, once)
// ---------------------------------------------------------------------------
__global__ __launch_bounds__(256)
void transpose_we_kernel(const float* __restrict__ We,
                         unsigned short* __restrict__ Wt) {
  __bf16* wt = (__bf16*)Wt;
  int i = blockIdx.x * 256 + threadIdx.x;           // grid covers H*ENC
  int n = i >> 10;                                  // 0..511
  int k = i & 1023;                                 // 0..1023
  wt[i] = (__bf16)We[(size_t)k * HSZ + n];
}

// ---------------------------------------------------------------------------
// K0: embedding gather + attention query q = dec @ Wd + bd   (64 x 512, tiny)
// ---------------------------------------------------------------------------
__global__ __launch_bounds__(512)
void prep_kernel(const int* __restrict__ ids, const float* __restrict__ hidden,
                 const float* __restrict__ emb, const float* __restrict__ Wd,
                 const float* __restrict__ bd, float* __restrict__ embedded,
                 float* __restrict__ q) {
  int b = blockIdx.x, h = threadIdx.x;
  __shared__ float dec_s[HSZ];
  dec_s[h] = hidden[(size_t)BSZ * HSZ + (size_t)b * HSZ + h];   // hidden[-1]
  embedded[(size_t)b * ESZ + h] = emb[(size_t)ids[b] * ESZ + h];
  __syncthreads();
  float acc = bd[h];
  for (int k = 0; k < HSZ; ++k)
    acc = fmaf(dec_s[k], Wd[(size_t)k * HSZ + h], acc);
  q[(size_t)b * HSZ + h] = acc;
}

// ---------------------------------------------------------------------------
// K1: scores[b,s] = att_v . tanh(enc[b,s,:]@We + be + q[b,:])  via WMMA bf16
// grid = 512 tiles of 16 (b,s) rows; 8 waves split the 32 N-tiles of H.
// ---------------------------------------------------------------------------
__global__ __launch_bounds__(256)
void attn_scores_kernel(const float* __restrict__ enc,
                        const unsigned short* __restrict__ Wt,  // [H][ENC] bf16
                        const float* __restrict__ be, const float* __restrict__ q,
                        const float* __restrict__ vvec, const int* __restrict__ mask,
                        float* __restrict__ scores) {
  int tile = blockIdx.x;              // 8 tiles per batch row (S=128)
  int b    = tile >> 3;
  int s0   = (tile & 7) * 16;
  int wave = threadIdx.x >> 5;
  int lane = threadIdx.x & 31;

  __shared__ float score_s[16];
  if (threadIdx.x < 16) score_s[threadIdx.x] = 0.f;
  __syncthreads();

  const float* Arow = enc + ((size_t)b * SSZ + s0) * ENC;

  float psum[8];
#pragma unroll
  for (int r = 0; r < 8; ++r) psum[r] = 0.f;

  for (int nt = wave; nt < 32; nt += 8) {
    int n0 = nt * 16;
    v8f c = {};
    for (int k0 = 0; k0 < ENC; k0 += 32) {
      v16bf a  = load_a_f32(Arow, ENC, k0);
      v16bf bm = load_b_bf16rows(Wt, ENC, k0, n0);   // B[k][n] = Wt[n][k]
      c = __builtin_amdgcn_wmma_f32_16x16x32_bf16(false, a, false, bm,
                                                  (short)0, c, false, false);
    }
    int   n   = n0 + (lane & 15);
    float add = be[n] + q[(size_t)b * HSZ + n];
    float vn  = vvec[n];
#pragma unroll
    for (int r = 0; r < 8; ++r)
      psum[r] += tanhf(c[r] + add) * vn;
  }

  // reduce the 16 lanes holding the same row m, then combine across waves
#pragma unroll
  for (int r = 0; r < 8; ++r) {
    float p = psum[r];
    for (int off = 8; off >= 1; off >>= 1)
      p += __shfl_xor(p, off, 16);
    if ((lane & 15) == 0) {
      int m = r + ((lane >> 4) << 3);
      atomicAdd(&score_s[m], p);
    }
  }
  __syncthreads();
  if (threadIdx.x < 16) {
    int   s  = s0 + threadIdx.x;
    float sc = score_s[threadIdx.x];
    scores[(size_t)b * SSZ + s] = (mask[(size_t)b * SSZ + s] > 0) ? sc : -1e9f;
  }
}

// ---------------------------------------------------------------------------
// K2: softmax over S + context = attn @ enc   (one block per batch row)
// ---------------------------------------------------------------------------
__global__ __launch_bounds__(128)
void softmax_ctx_kernel(const float* __restrict__ scores,
                        const float* __restrict__ enc,
                        float* __restrict__ attn_out,
                        float* __restrict__ context) {
  int b = blockIdx.x, t = threadIdx.x;            // 128 threads
  __shared__ float red[SSZ];
  __shared__ float attn_s[SSZ];
  float s = scores[(size_t)b * SSZ + t];
  red[t] = s; __syncthreads();
  for (int off = 64; off >= 1; off >>= 1) {
    if (t < off) red[t] = fmaxf(red[t], red[t + off]);
    __syncthreads();
  }
  float mx = red[0]; __syncthreads();
  float ex = __expf(s - mx);
  red[t] = ex; __syncthreads();
  for (int off = 64; off >= 1; off >>= 1) {
    if (t < off) red[t] += red[t + off];
    __syncthreads();
  }
  float a = ex / red[0];
  attn_s[t] = a;
  attn_out[(size_t)b * SSZ + t] = a;
  __syncthreads();
  for (int e0 = 0; e0 < ENC; e0 += 128) {
    int col = e0 + t;
    float acc = 0.f;
    for (int s2 = 0; s2 < SSZ; ++s2)
      acc = fmaf(attn_s[s2], enc[((size_t)b * SSZ + s2) * ENC + col], acc);
    context[(size_t)b * ENC + col] = acc;
  }
}

// ---------------------------------------------------------------------------
// K3: GRU layer 0 (x = [embedded, context], K=1536; hprev = hidden[0])
// ---------------------------------------------------------------------------
__global__ __launch_bounds__(512)
void gru0_kernel(const float* __restrict__ embedded, const float* __restrict__ context,
                 const float* __restrict__ hidden,
                 const float* __restrict__ w_ih, const float* __restrict__ w_hh,
                 const float* __restrict__ b_ih, const float* __restrict__ b_hh,
                 float* __restrict__ h0_out) {
  int b = blockIdx.x, h = threadIdx.x;
  __shared__ float x_s[ESZ + ENC];
  __shared__ float hp_s[HSZ];
  x_s[h]             = embedded[(size_t)b * ESZ + h];
  x_s[ESZ + h]       = context[(size_t)b * ENC + h];
  x_s[ESZ + HSZ + h] = context[(size_t)b * ENC + HSZ + h];
  hp_s[h]            = hidden[(size_t)b * HSZ + h];       // layer 0
  __syncthreads();
  float gi[3], gh[3];
#pragma unroll
  for (int g = 0; g < 3; ++g) {
    int j = g * HSZ + h;
    float acc = b_ih[j];
    const float* wr = w_ih + (size_t)j * (ESZ + ENC);
    for (int k = 0; k < ESZ + ENC; ++k) acc = fmaf(x_s[k], wr[k], acc);
    gi[g] = acc;
    float acc2 = b_hh[j];
    const float* wr2 = w_hh + (size_t)j * HSZ;
    for (int k = 0; k < HSZ; ++k) acc2 = fmaf(hp_s[k], wr2[k], acc2);
    gh[g] = acc2;
  }
  float r = 1.f / (1.f + __expf(-(gi[0] + gh[0])));
  float z = 1.f / (1.f + __expf(-(gi[1] + gh[1])));
  float n = tanhf(gi[2] + r * gh[2]);
  h0_out[(size_t)b * HSZ + h] = (1.f - z) * n + z * hp_s[h];
}

// ---------------------------------------------------------------------------
// K4: GRU layer 1 (x = h0, K=512) + emit out_in = [h1, context, embedded]
//     directly in bf16 (fc A-operand needs no per-tile conversion).
// ---------------------------------------------------------------------------
__global__ __launch_bounds__(512)
void gru1_kernel(const float* __restrict__ h0, const float* __restrict__ hidden,
                 const float* __restrict__ embedded, const float* __restrict__ context,
                 const float* __restrict__ w_ih, const float* __restrict__ w_hh,
                 const float* __restrict__ b_ih, const float* __restrict__ b_hh,
                 float* __restrict__ h1_out, unsigned short* __restrict__ out_in_bf) {
  int b = blockIdx.x, h = threadIdx.x;
  __shared__ float x_s[HSZ];
  __shared__ float hp_s[HSZ];
  x_s[h]  = h0[(size_t)b * HSZ + h];
  hp_s[h] = hidden[(size_t)BSZ * HSZ + (size_t)b * HSZ + h];  // layer 1
  __syncthreads();
  float gi[3], gh[3];
#pragma unroll
  for (int g = 0; g < 3; ++g) {
    int j = g * HSZ + h;
    float acc = b_ih[j];
    const float* wr = w_ih + (size_t)j * HSZ;
    for (int k = 0; k < HSZ; ++k) acc = fmaf(x_s[k], wr[k], acc);
    gi[g] = acc;
    float acc2 = b_hh[j];
    const float* wr2 = w_hh + (size_t)j * HSZ;
    for (int k = 0; k < HSZ; ++k) acc2 = fmaf(hp_s[k], wr2[k], acc2);
    gh[g] = acc2;
  }
  float r  = 1.f / (1.f + __expf(-(gi[0] + gh[0])));
  float z  = 1.f / (1.f + __expf(-(gi[1] + gh[1])));
  float n  = tanhf(gi[2] + r * gh[2]);
  float h1 = (1.f - z) * n + z * hp_s[h];
  h1_out[(size_t)b * HSZ + h] = h1;
  __bf16* oi = (__bf16*)out_in_bf + (size_t)b * KOUT;
  oi[h]             = (__bf16)h1;
  oi[HSZ + h]       = (__bf16)context[(size_t)b * ENC + h];
  oi[HSZ + HSZ + h] = (__bf16)context[(size_t)b * ENC + HSZ + h];
  oi[HSZ + ENC + h] = (__bf16)embedded[(size_t)b * ESZ + h];
}

// ---------------------------------------------------------------------------
// K5: output = out_in @ fc_w^T + fc_b   (64 x 2048 x 32000) via WMMA bf16.
// One block (4 waves = 4 M-tiles) per N-tile of 16 vocab rows.  fc_w B-tiles
// (16 rows x 32 K of f32, 2KB) are double-buffered through LDS with
// GLOBAL_LOAD_ASYNC_TO_LDS_B128 (ASYNCcnt), so the 262 MB stream is fetched
// from HBM exactly once per block and shared by the 4 consumer waves.
// ---------------------------------------------------------------------------
__global__ __launch_bounds__(128)
void fc_wmma_kernel(const unsigned short* __restrict__ out_in_bf,
                    const float* __restrict__ fc_w,
                    const float* __restrict__ fc_b, float* __restrict__ out) {
  int ntile = blockIdx.x;                 // 0..1999
  int mtile = threadIdx.x >> 5;           // 0..3
  int lane  = threadIdx.x & 31;
  int n0    = ntile * 16;
  int m0    = mtile * 16;

  __shared__ float bstage[2][16][32];     // 2 x 2KB double buffer

  // cooperative stage of one 16x32 f32 chunk: 128 threads x 16B each
  int   srow = threadIdx.x >> 3;                       // 0..15
  int   scol = (threadIdx.x & 7) << 2;                 // 0,4,..,28
  const float* gbase = fc_w + (size_t)(n0 + srow) * KOUT + scol;

#define STAGE(bufidx, k0)                                                     \
  do {                                                                        \
    const float* gsrc = gbase + (k0);                                         \
    float*       ldst = &bstage[(bufidx)][srow][scol];                        \
    __builtin_prefetch(gsrc + 256, 0, 0);                                     \
    if (HAVE_ASYNC_LDS) {                                                     \
      ASYNC_COPY16(gsrc, ldst);                                               \
    } else {                                                                  \
      *(float4*)ldst = *(const float4*)gsrc;                                  \
    }                                                                         \
  } while (0)

  const unsigned short* Arow = out_in_bf + (size_t)m0 * KOUT;

  v8f c = {};
  STAGE(0, 0);                                        // prologue: buffer 0
  for (int s = 0; s < KOUT / 32; ++s) {               // 64 K-steps
    int k0 = s * 32;
    if (s + 1 < KOUT / 32) {                          // uniform branch
      STAGE((s + 1) & 1, k0 + 32);
      WAIT_ASYNC_N(1);                                // current buffer landed
    } else {
      WAIT_ASYNC_N(0);                                // drain for last step
    }
    __syncthreads();

    // consume bstage[s&1]
    v16bf a = load_a_bf16(Arow, KOUT, k0);
    const float* brow = &bstage[s & 1][lane & 15][(lane < 16) ? 0 : 16];
    v16bf bm;
#pragma unroll
    for (int e = 0; e < 16; ++e) bm[e] = (__bf16)brow[e];
    c = __builtin_amdgcn_wmma_f32_16x16x32_bf16(false, a, false, bm,
                                                (short)0, c, false, false);
    __syncthreads();                                  // before buffer reuse
  }

  int   n    = n0 + (lane & 15);
  float bias = fc_b[n];
  int   mh   = m0 + ((lane >> 4) << 3);
#pragma unroll
  for (int r = 0; r < 8; ++r)
    out[(size_t)(mh + r) * VSZ + n] = c[r] + bias;
}

// ---------------------------------------------------------------------------
// Host-side launcher
// ---------------------------------------------------------------------------
extern "C" void kernel_launch(void* const* d_in, const int* in_sizes, int n_in,
                              void* d_out, int out_size, void* d_ws, size_t ws_size,
                              hipStream_t stream) {
  const int*   ids    = (const int*)  d_in[0];
  const float* hidden = (const float*)d_in[1];
  const float* enc    = (const float*)d_in[2];
  const int*   mask   = (const int*)  d_in[3];
  const float* emb    = (const float*)d_in[4];
  const float* att_We = (const float*)d_in[5];
  const float* att_be = (const float*)d_in[6];
  const float* att_Wd = (const float*)d_in[7];
  const float* att_bd = (const float*)d_in[8];
  const float* att_v  = (const float*)d_in[9];
  const float* w_ih0  = (const float*)d_in[10];
  const float* w_hh0  = (const float*)d_in[11];
  const float* b_ih0  = (const float*)d_in[12];
  const float* b_hh0  = (const float*)d_in[13];
  const float* w_ih1  = (const float*)d_in[14];
  const float* w_hh1  = (const float*)d_in[15];
  const float* b_ih1  = (const float*)d_in[16];
  const float* b_hh1  = (const float*)d_in[17];
  const float* fc_w   = (const float*)d_in[18];
  const float* fc_b   = (const float*)d_in[19];

  float* out    = (float*)d_out;                        // [B, V]
  float* h0_out = out + (size_t)BSZ * VSZ;              // new_hidden[0]
  float* h1_out = h0_out + (size_t)BSZ * HSZ;           // new_hidden[1]
  float* attn   = h1_out + (size_t)BSZ * HSZ;           // [B, S]

  float* ws       = (float*)d_ws;
  float* embedded = ws;                                 // B*E   = 32768 f32
  float* q        = embedded + (size_t)BSZ * ESZ;       // B*H   = 32768 f32
  float* scores   = q + (size_t)BSZ * HSZ;              // B*S   =  8192 f32
  float* context  = scores + (size_t)BSZ * SSZ;         // B*ENC = 65536 f32
  unsigned short* we_t      = (unsigned short*)(context + (size_t)BSZ * ENC);
  unsigned short* out_in_bf = we_t + (size_t)HSZ * ENC; // 64*2048 bf16

  transpose_we_kernel<<<(HSZ * ENC) / 256, 256, 0, stream>>>(att_We, we_t);
  prep_kernel<<<BSZ, HSZ, 0, stream>>>(ids, hidden, emb, att_Wd, att_bd,
                                       embedded, q);
  attn_scores_kernel<<<(BSZ * SSZ) / 16, 256, 0, stream>>>(
      enc, we_t, att_be, q, att_v, mask, scores);
  softmax_ctx_kernel<<<BSZ, SSZ, 0, stream>>>(scores, enc, attn, context);
  gru0_kernel<<<BSZ, HSZ, 0, stream>>>(embedded, context, hidden,
                                       w_ih0, w_hh0, b_ih0, b_hh0, h0_out);
  gru1_kernel<<<BSZ, HSZ, 0, stream>>>(h0_out, hidden, embedded, context,
                                       w_ih1, w_hh1, b_ih1, b_hh1, h1_out,
                                       out_in_bf);
  fc_wmma_kernel<<<VSZ / 16, 128, 0, stream>>>(out_in_bf, fc_w, fc_b, out);
}
